// Mamba3Mixer_21577915695231
// MI455X (gfx1250) — matrix operand order repaired and checked by reading, not compiled, not run on hardware
//
#include <hip/hip_runtime.h>
#include <hip/hip_bf16.h>
#include <math.h>

typedef __attribute__((ext_vector_type(16))) __bf16 v16bf;
typedef __attribute__((ext_vector_type(8)))  __bf16 v8bf;
typedef __attribute__((ext_vector_type(8)))  float  v8f;
typedef __attribute__((ext_vector_type(4)))  unsigned int u32x4;
typedef __attribute__((ext_vector_type(8)))  int i32x8;

#define D_MODEL   1024
#define D_INNER   2048
#define N_HEADS   32
#define HEAD_DIM  64
#define D_STATE   128
#define CHUNK     256
#define SEQ       2048
#define BATCH     2
#define NTOK      (BATCH*SEQ)          // 4096
#define PROJ_DIM  14400
#define ZOFF      0
#define XOFF      2048
#define BOFF      4096
#define COFF      8192
#define DTOFF     12288
#define THOFF     12352
#define LOGEPS    (-13.815510558f)     // log(1e-6)

// ---- bf16 vector load helpers matching WMMA per-lane layouts ---------------
__device__ inline v16bf load_a16(const __bf16* p) {   // A: 16 contiguous K
  v8bf lo = *(const v8bf*)p;
  v8bf hi = *(const v8bf*)(p + 8);
  return __builtin_shufflevector(lo, hi, 0,1,2,3,4,5,6,7,8,9,10,11,12,13,14,15);
}
__device__ inline v16bf load_b16(const __bf16* p) {   // B: 16 contiguous K
  v8bf lo = *(const v8bf*)p;
  v8bf hi = *(const v8bf*)(p + 8);
  return __builtin_shufflevector(lo, hi, 0,1,2,3,4,5,6,7,8,9,10,11,12,13,14,15);
}

// ---- Tensor Data Mover: 2-D tile (rows x tile_k bf16) global -> LDS --------
// D# per CDNA5 ISA: group0 {count, lds_addr, global_addr,
// type=2}, group1 {data_size=2B, dims, tile, stride}. Tracked by TENSORcnt.
__device__ inline void tdm_load_2d_bf16(unsigned int lds_addr, const void* gaddr,
                                        int row_len, int rows, int tile_k)
{
  unsigned long long ga = (unsigned long long)(uintptr_t)gaddr;
  u32x4 g0;
  g0[0] = 1u;                                        // count=1, user mode
  g0[1] = lds_addr;                                  // LDS byte address
  g0[2] = (unsigned int)(ga & 0xffffffffu);          // global_addr[31:0]
  g0[3] = (unsigned int)((ga >> 32) & 0x01ffffffu)   // global_addr[56:32]
        | (2u << 30);                                // type=2 (image)
  i32x8 g1;
  g1[0] = (1 << 16);                                 // data_size=1 -> 2 bytes
  g1[1] = (int)((row_len & 0xffff) << 16);           // tensor_dim0[15:0]
  g1[2] = (int)(((unsigned)row_len >> 16)            // tensor_dim0[31:16]
        | ((unsigned)(rows & 0xffff) << 16));        // tensor_dim1[15:0]
  g1[3] = (int)((tile_k & 0xffff) << 16);            // dim1 hi=0 | tile_dim0
  g1[4] = rows & 0xffff;                             // tile_dim1 (tile_dim2=0)
  g1[5] = row_len;                                   // tensor_dim0_stride lo
  g1[6] = 0;                                         // stride hi | dim1_stride lo
  g1[7] = 0;
  asm volatile("tensor_load_to_lds %0, %1" :: "s"(g0), "s"(g1) : "memory");
}

// ---------------------------------------------------------------------------
// C = A * B^T, bf16 operands, fp32 accumulate.
// A: [M x K] bf16 row-major, B: [N x K] bf16 row-major, C: [M x N] fp32.
// Block: 256 thr / 8 waves, tile 128(M) x 64(N); each wave 16x64 (4 acc).
// B block-tile [64 x 32] staged in LDS by the TDM, double buffered.
// ---------------------------------------------------------------------------
__global__ __launch_bounds__(256) void gemm_tdm_bf16(
    const __bf16* __restrict__ A, const __bf16* __restrict__ B,
    float* __restrict__ C, int M, int N, int K)
{
  const int wave  = threadIdx.x >> 5;
  const int lane  = threadIdx.x & 31;
  const int lrow  = lane & 15;
  const int khalf = lane >> 4;
  const int n0 = blockIdx.x * 64;
  const int m0 = blockIdx.y * 128 + wave * 16;

  __shared__ __bf16 sB[2][64][32];                  // 4 KB per buffer

  const __bf16* ap0 = A + (size_t)(m0 + lrow) * K;
  const __bf16* bn0 = B + (size_t)n0 * K;

  if (wave == 0)
    tdm_load_2d_bf16((unsigned int)(uintptr_t)&sB[0][0][0], bn0, K, 64, 32);

  v8f acc[4] = {};
  for (int k0 = 0; k0 < K; k0 += 32) {
    const int buf = (k0 >> 5) & 1;
    if (wave == 0) __builtin_amdgcn_s_wait_tensorcnt(0);
    __syncthreads();                                 // TDM tile visible to WGP
    if (wave == 0 && k0 + 32 < K)                    // prefetch next tile
      tdm_load_2d_bf16((unsigned int)(uintptr_t)&sB[buf ^ 1][0][0],
                       bn0 + (k0 + 32), K, 64, 32);
    __builtin_prefetch(ap0 + k0 + 256, 0, 0);        // global_prefetch_b8
    const v16bf a = load_a16(ap0 + k0 + khalf * 16);
#pragma unroll
    for (int nt = 0; nt < 4; ++nt) {
      const v16bf bm = load_b16(&sB[buf][nt * 16 + lrow][khalf * 16]);
      acc[nt] = __builtin_amdgcn_wmma_f32_16x16x32_bf16(false, a, false, bm,
                                                        (short)0, acc[nt], false, false);
    }
    __syncthreads();                                 // done reading this buffer
  }
#pragma unroll
  for (int nt = 0; nt < 4; ++nt)
#pragma unroll
    for (int v = 0; v < 8; ++v)
      C[(size_t)(m0 + v + 8 * khalf) * N + n0 + nt * 16 + lrow] = acc[nt][v];
}

// ---------------------------------------------------------------------------
__global__ void cvt_bf16_kernel(const float* __restrict__ s, __bf16* __restrict__ d)
{
  const int i = (blockIdx.x * 256 + threadIdx.x) * 4;
#pragma unroll
  for (int q = 0; q < 4; ++q) d[i + q] = (__bf16)s[i + q];
}

__global__ void head_a_kernel(const float* __restrict__ A_log, float* __restrict__ aH)
{
  int h = threadIdx.x;
  if (h < N_HEADS) {
    float s = 0.f;
    for (int n = 0; n < D_STATE; ++n) s += expf(A_log[h * D_STATE + n]);
    aH[h] = s * (1.f / D_STATE);
  }
}

__device__ inline float softplus_f(float x) {
  return (x > 20.f) ? x : log1pf(expf(x));
}

// ---------------------------------------------------------------------------
// Per-token prep: gates, RMSNorm(B/C)+bias (fp32, pre-rotary), theta*dt1,
// and x restrided to bf16 [b*32+h][p][s] for contiguous-K WMMA feeds.
// ---------------------------------------------------------------------------
__global__ __launch_bounds__(256) void prep_kernel(
    const float* __restrict__ proj, const float* __restrict__ aH,
    const float* __restrict__ Bw, const float* __restrict__ Cw,
    const float* __restrict__ Bb, const float* __restrict__ Cb,
    const float* __restrict__ dtb,
    float* __restrict__ BmW, float* __restrict__ CmW, float* __restrict__ thdt,
    float* __restrict__ logaO, float* __restrict__ betaO, float* __restrict__ gamO,
    __bf16* __restrict__ xbf)
{
  const int m    = blockIdx.x;
  const int b    = m / SEQ;
  const int s    = m % SEQ;
  const int wave = threadIdx.x >> 5;
  const int lane = threadIdx.x & 31;
  const float* pr = proj + (size_t)m * PROJ_DIM;

  for (int hg = 0; hg < 4; ++hg) {
    const int h  = hg * 8 + wave;
    const int bh = b * N_HEADS + h;
    const float dt1 = softplus_f(pr[DTOFF + h] + dtb[h]);
    const float dt2 = softplus_f(pr[DTOFF + N_HEADS + h] + dtb[N_HEADS + h]);
    const float la  = -dt1 * aH[h];
    if (lane == 0) {
      const float gam = 0.5f * dt2;
      logaO[(size_t)bh * SEQ + s] = la;
      gamO [(size_t)bh * SEQ + s] = gam;
      betaO[(size_t)bh * SEQ + s] = gam * expf(la);
    }
    float vB[4], vC[4];
    float ssB = 0.f, ssC = 0.f;
#pragma unroll
    for (int q = 0; q < 4; ++q) {
      int n = lane + 32 * q;
      vB[q] = pr[BOFF + h * D_STATE + n];
      vC[q] = pr[COFF + h * D_STATE + n];
      ssB += vB[q] * vB[q];
      ssC += vC[q] * vC[q];
    }
    for (int d = 16; d > 0; d >>= 1) {
      ssB += __shfl_xor(ssB, d, 32);
      ssC += __shfl_xor(ssC, d, 32);
    }
    const float rB = rsqrtf(ssB * (1.f / D_STATE) + 1e-6f);
    const float rC = rsqrtf(ssC * (1.f / D_STATE) + 1e-6f);
    float* Bo = BmW + ((size_t)bh * SEQ + s) * D_STATE;
    float* Co = CmW + ((size_t)bh * SEQ + s) * D_STATE;
#pragma unroll
    for (int q = 0; q < 4; ++q) {
      int n = lane + 32 * q;
      Bo[n] = vB[q] * rB * Bw[n] * Bb[h * D_STATE + n];
      Co[n] = vC[q] * rC * Cw[n] * Cb[h * D_STATE + n];
    }
#pragma unroll
    for (int q = 0; q < 2; ++q) {
      int k = lane + 32 * q;
      thdt[((size_t)bh * (D_STATE / 2) + k) * SEQ + s] =
          pr[THOFF + h * (D_STATE / 2) + k] * dt1;
      // x restride: [bh][p][s]
      xbf[((size_t)bh * HEAD_DIM + k) * SEQ + s] =
          (__bf16)pr[XOFF + h * HEAD_DIM + k];
    }
  }
}

// ---------------------------------------------------------------------------
// freqs = cumsum_t(theta*dt1); rotary(B), rotary(C) -> bf16 outputs.
// One wave per (b,h,k) row; lane handles 64 contiguous t (shfl_up prefix).
// ---------------------------------------------------------------------------
__global__ __launch_bounds__(256) void cumsum_rotary_kernel(
    const float* __restrict__ thdt,
    const float* __restrict__ BmW, const float* __restrict__ CmW,
    __bf16* __restrict__ BmH, __bf16* __restrict__ CmH)
{
  const int wave = threadIdx.x >> 5;
  const int lane = threadIdx.x & 31;
  const int r  = blockIdx.x * 8 + wave;
  const int bh = r / (D_STATE / 2);
  const int k  = r % (D_STATE / 2);
  const float* tr = thdt + (size_t)r * SEQ;
  const float* Bp = BmW + (size_t)bh * SEQ * D_STATE;
  const float* Cp = CmW + (size_t)bh * SEQ * D_STATE;
  __bf16* Bo = BmH + (size_t)bh * SEQ * D_STATE;
  __bf16* Co = CmH + (size_t)bh * SEQ * D_STATE;

  const int t0 = lane * 64;
  float lsum = 0.f;
  for (int i = 0; i < 64; ++i) lsum += tr[t0 + i];
  float xs = lsum;
  for (int d = 1; d < 32; d <<= 1) {
    float y = __shfl_up(xs, (unsigned)d, 32);
    if (lane >= d) xs += y;
  }
  float run = xs - lsum;                   // exclusive prefix
  for (int i = 0; i < 64; ++i) {
    const int t = t0 + i;
    run += tr[t];
    const float cs = cosf(run), sn = sinf(run);
    const float b1 = Bp[(size_t)t * D_STATE + 2 * k];
    const float b2 = Bp[(size_t)t * D_STATE + 2 * k + 1];
    Bo[(size_t)t * D_STATE + 2 * k]     = (__bf16)(b1 * cs - b2 * sn);
    Bo[(size_t)t * D_STATE + 2 * k + 1] = (__bf16)(b1 * sn + b2 * cs);
    const float c1 = Cp[(size_t)t * D_STATE + 2 * k];
    const float c2 = Cp[(size_t)t * D_STATE + 2 * k + 1];
    Co[(size_t)t * D_STATE + 2 * k]     = (__bf16)(c1 * cs - c2 * sn);
    Co[(size_t)t * D_STATE + 2 * k + 1] = (__bf16)(c1 * sn + c2 * cs);
  }
}

// ---------------------------------------------------------------------------
// Chunked trapezoidal scan. One block per (b,head); 8 waves; 8 chunks serial.
// State h[128x64] fp32 in LDS. All matmuls v_wmma_f32_16x16x32_bf16 with
// bf16 operand arrays; masked M tile re-layouted through per-wave LDS tile.
// Writes yzbf = (y * silu(z)) in bf16 [4096 x 2048] for the output GEMM.
// ---------------------------------------------------------------------------
__global__ __launch_bounds__(256) void chunk_scan_kernel(
    const float* __restrict__ proj,
    const __bf16* __restrict__ BmH, const __bf16* __restrict__ CmH,
    const __bf16* __restrict__ xbf,
    const float* __restrict__ logaA, const float* __restrict__ betaA,
    const float* __restrict__ gamA, __bf16* __restrict__ yzbf)
{
  const int bh   = blockIdx.x;
  const int b    = bh >> 5;
  const int h    = bh & 31;
  const int tid  = threadIdx.x;
  const int wave = tid >> 5;
  const int lane = tid & 31;
  const int lrow = lane & 15;
  const int khalf = lane >> 4;

  __shared__ float sLac[CHUNK], sBeta[CHUNK], sGam[CHUNK], sBN[CHUNK], sW[CHUNK], sEl[CHUNK];
  __shared__ float sWsum[8];
  __shared__ float sH[D_STATE][HEAD_DIM];          // recurrent state (fp32)
  __shared__ __bf16 sM[8][16][32];                 // per-wave M staging

  for (int i = tid; i < D_STATE * HEAD_DIM; i += 256)
    sH[i / HEAD_DIM][i % HEAD_DIM] = 0.f;

  const __bf16* Bh  = BmH + (size_t)bh * SEQ * D_STATE;
  const __bf16* Chp = CmH + (size_t)bh * SEQ * D_STATE;
  const __bf16* Xh  = xbf + (size_t)bh * HEAD_DIM * SEQ;
  const float* lg  = logaA + (size_t)bh * SEQ;
  const float* bet = betaA + (size_t)bh * SEQ;
  const float* gam = gamA  + (size_t)bh * SEQ;
  __syncthreads();

  for (int c = 0; c < SEQ / CHUNK; ++c) {
    const int gc0 = c * CHUNK;

    // ---- phase 1: inclusive scan of log alpha over the chunk -------------
    const float lal = fmaxf(lg[gc0 + tid], LOGEPS);
    float xs = lal;
    for (int d = 1; d < 32; d <<= 1) {
      float y = __shfl_up(xs, (unsigned)d, 32);
      if (lane >= d) xs += y;
    }
    if (lane == 31) sWsum[wave] = xs;
    sBeta[tid] = bet[gc0 + tid];
    sGam[tid]  = gam[gc0 + tid];
    __syncthreads();
    float off = 0.f;
    for (int w2 = 0; w2 < wave; ++w2) off += sWsum[w2];
    sLac[tid] = xs + off;
    __syncthreads();
    const float lacLast = sLac[CHUNK - 1];
    sBN[tid] = (tid < CHUNK - 1) ? sBeta[tid + 1] : 0.f;
    sW[tid]  = expf(lacLast - sLac[tid]) * sGam[tid] +
               ((tid < CHUNK - 1) ? expf(lacLast - sLac[tid + 1]) * sBeta[tid + 1] : 0.f);
    sEl[tid] = expf(sLac[tid]);
    __syncthreads();

    // ---- phase 2: y = (M ⊙ masks) @ x  +  exp(lac)*(C @ h) ---------------
    for (int tt = 0; tt < 2; ++tt) {
      const int T   = wave * 2 + tt;               // t-tile index 0..15
      const int gt0 = gc0 + T * 16;
      v8f yacc[4] = {};
      v8f cacc[4] = {};

      for (int j = 0; j <= T / 2; ++j) {           // s-tile pairs (K=32)
        for (int half = 0; half < 2; ++half) {
          const int S = 2 * j + half;
          if (S > T) {
#pragma unroll
            for (int v = 0; v < 8; ++v)
              sM[wave][v + 8 * khalf][half * 16 + lrow] = (__bf16)0.f;
          } else {
            const int gs0 = gc0 + S * 16;
            v8f macc = {};
            for (int k0 = 0; k0 < D_STATE; k0 += 32) {   // CB = C . B^T
              const v16bf a  = load_a16(Chp + (size_t)(gt0 + lrow) * D_STATE + k0 + khalf * 16);
              const v16bf bm = load_b16(Bh  + (size_t)(gs0 + lrow) * D_STATE + k0 + khalf * 16);
              macc = __builtin_amdgcn_wmma_f32_16x16x32_bf16(false, a, false, bm,
                                                             (short)0, macc, false, false);
            }
            const int scol = S * 16 + lrow;
#pragma unroll
            for (int v = 0; v < 8; ++v) {
              const int trow = T * 16 + v + 8 * khalf;
              float val = 0.f;
              if (scol <= trow) {
                const float coef = (scol == trow) ? sGam[trow] : sBN[scol];
                val = macc[v] * expf(sLac[trow] - sLac[scol]) * coef;
              }
              sM[wave][v + 8 * khalf][half * 16 + lrow] = (__bf16)val;
            }
          }
        }
        __builtin_amdgcn_wave_barrier();           // D-layout write -> A read
        const v16bf aM = load_a16(&sM[wave][lrow][khalf * 16]);
        __builtin_amdgcn_wave_barrier();
#pragma unroll
        for (int pt = 0; pt < 4; ++pt) {           // y += M @ x
          const int p = pt * 16 + lrow;
          const v16bf bx = load_b16(Xh + (size_t)p * SEQ + gc0 + 32 * j + khalf * 16);
          yacc[pt] = __builtin_amdgcn_wmma_f32_16x16x32_bf16(false, aM, false, bx,
                                                             (short)0, yacc[pt], false, false);
        }
      }

      // inter-chunk term: C @ h  (K = 128 state)
      for (int k0 = 0; k0 < D_STATE; k0 += 32) {
        const v16bf a = load_a16(Chp + (size_t)(gt0 + lrow) * D_STATE + k0 + khalf * 16);
#pragma unroll
        for (int pt = 0; pt < 4; ++pt) {
          v16bf bhm;
#pragma unroll
          for (int q = 0; q < 16; ++q)
            bhm[q] = (__bf16)sH[k0 + khalf * 16 + q][pt * 16 + lrow];
          cacc[pt] = __builtin_amdgcn_wmma_f32_16x16x32_bf16(false, a, false, bhm,
                                                             (short)0, cacc[pt], false, false);
        }
      }

      // combine, gate with silu(z), emit yz (bf16)
#pragma unroll
      for (int pt = 0; pt < 4; ++pt) {
#pragma unroll
        for (int v = 0; v < 8; ++v) {
          const int tloc = T * 16 + v + 8 * khalf;
          const int gm   = b * SEQ + gc0 + tloc;
          const int col  = h * HEAD_DIM + pt * 16 + lrow;
          const float yv = yacc[pt][v] + sEl[tloc] * cacc[pt][v];
          const float z  = proj[(size_t)gm * PROJ_DIM + ZOFF + col];
          const float sz = z / (1.f + expf(-z));
          yzbf[(size_t)gm * D_INNER + col] = (__bf16)(yv * sz);
        }
      }
    }

    __syncthreads();   // all waves finished reading sH for this chunk

    // ---- phase 3: h = exp(lac[-1])*h + B^T @ (w ⊙ x) ---------------------
    {
      const int n0 = wave * 16;
      v8f hacc[4] = {};
      for (int k0 = 0; k0 < CHUNK; k0 += 32) {
        v16bf aB;
#pragma unroll
        for (int q = 0; q < 8; ++q) {
          aB[q]     = Bh[(size_t)(gc0 + k0 + khalf * 16 + q) * D_STATE + n0 + lrow];
          aB[q + 8] = Bh[(size_t)(gc0 + k0 + khalf * 16 + 8 + q) * D_STATE + n0 + lrow];
        }
#pragma unroll
        for (int pt = 0; pt < 4; ++pt) {
          const int p = pt * 16 + lrow;
          const __bf16* xr = Xh + (size_t)p * SEQ + gc0 + k0 + khalf * 16;
          v16bf bxw;
#pragma unroll
          for (int q = 0; q < 16; ++q)
            bxw[q] = (__bf16)(sW[k0 + khalf * 16 + q] * (float)xr[q]);
          hacc[pt] = __builtin_amdgcn_wmma_f32_16x16x32_bf16(false, aB, false, bxw,
                                                             (short)0, hacc[pt], false, false);
        }
      }
      const float dec = expf(lacLast);
#pragma unroll
      for (int pt = 0; pt < 4; ++pt) {
#pragma unroll
        for (int v = 0; v < 8; ++v) {
          const int n = n0 + v + 8 * khalf;
          const int p = pt * 16 + lrow;
          sH[n][p] = dec * sH[n][p] + hacc[pt][v];
        }
      }
    }
    __syncthreads();
  }
}

// ---------------------------------------------------------------------------
extern "C" void kernel_launch(void* const* d_in, const int* in_sizes, int n_in,
                              void* d_out, int out_size, void* d_ws, size_t ws_size,
                              hipStream_t stream)
{
  const float* u      = (const float*)d_in[0];
  const float* W_in   = (const float*)d_in[1];
  const float* W_out  = (const float*)d_in[2];
  const float* A_log  = (const float*)d_in[3];
  const float* Bw     = (const float*)d_in[4];
  const float* Cw     = (const float*)d_in[5];
  const float* Bb     = (const float*)d_in[6];
  const float* Cb     = (const float*)d_in[7];
  const float* dtb    = (const float*)d_in[8];
  float* out = (float*)d_out;

  float* ws    = (float*)d_ws;
  float* proj  = ws;                                        // 4096*14400 f32
  float* aH    = proj  + (size_t)NTOK * PROJ_DIM;
  float* BmW   = aH    + 32;
  float* CmW   = BmW   + (size_t)BATCH * N_HEADS * SEQ * D_STATE;
  float* thdt  = CmW   + (size_t)BATCH * N_HEADS * SEQ * D_STATE;
  float* loga  = thdt  + (size_t)BATCH * N_HEADS * (D_STATE / 2) * SEQ;
  float* betaA = loga  + (size_t)BATCH * N_HEADS * SEQ;
  float* gamA  = betaA + (size_t)BATCH * N_HEADS * SEQ;
  __bf16* ubf   = (__bf16*)(gamA + (size_t)BATCH * N_HEADS * SEQ);
  __bf16* Winbf = ubf   + (size_t)NTOK * D_MODEL;
  __bf16* Woutbf= Winbf + (size_t)PROJ_DIM * D_MODEL;
  __bf16* BmH   = Woutbf+ (size_t)D_MODEL * D_INNER;
  __bf16* CmH   = BmH   + (size_t)BATCH * N_HEADS * SEQ * D_STATE;
  __bf16* xbf   = CmH   + (size_t)BATCH * N_HEADS * SEQ * D_STATE;
  __bf16* yzbf  = xbf   + (size_t)BATCH * N_HEADS * SEQ * HEAD_DIM;

  head_a_kernel<<<1, 32, 0, stream>>>(A_log, aH);

  cvt_bf16_kernel<<<(NTOK * D_MODEL) / 1024, 256, 0, stream>>>(u, ubf);
  cvt_bf16_kernel<<<(PROJ_DIM * D_MODEL) / 1024, 256, 0, stream>>>(W_in, Winbf);
  cvt_bf16_kernel<<<(D_MODEL * D_INNER) / 1024, 256, 0, stream>>>(W_out, Woutbf);

  dim3 g1(PROJ_DIM / 64, NTOK / 128);
  gemm_tdm_bf16<<<g1, 256, 0, stream>>>(ubf, Winbf, proj, NTOK, PROJ_DIM, D_MODEL);

  prep_kernel<<<NTOK, 256, 0, stream>>>(proj, aH, Bw, Cw, Bb, Cb, dtb,
                                        BmW, CmW, thdt, loga, betaA, gamA, xbf);

  cumsum_rotary_kernel<<<(BATCH * N_HEADS * (D_STATE / 2)) / 8, 256, 0, stream>>>(
      thdt, BmW, CmW, BmH, CmH);

  chunk_scan_kernel<<<BATCH * N_HEADS, 256, 0, stream>>>(
      proj, BmH, CmH, xbf, loga, betaA, gamA, yzbf);

  dim3 g2(D_MODEL / 64, NTOK / 128);
  gemm_tdm_bf16<<<g2, 256, 0, stream>>>(yzbf, Woutbf, out, NTOK, D_MODEL, D_INNER);
}
